// BDH_GPU_51135880626805
// MI455X (gfx1250) — matrix operand order repaired and checked by reading, not compiled, not run on hardware
//
#include <hip/hip_runtime.h>
#include <hip/hip_bf16.h>

// Problem constants (match reference)
#define BB   4
#define TT   2048
#define DD   256
#define HH   4
#define MM   2048      // N/H
#define NNN  8192
#define VV   256
#define LL   6
#define ROWS (BB * TT) // 8192

typedef __attribute__((ext_vector_type(16))) __bf16 v16bf;
typedef __attribute__((ext_vector_type(8)))  __bf16 v8bf;
typedef __attribute__((ext_vector_type(8)))  float  v8f;

// ---------------------------------------------------------------------------
// WMMA helpers (CDNA5 wave32, 16x16x32 bf16, f32 accumulate)
// ---------------------------------------------------------------------------
__device__ __forceinline__ v8f wmma_bf16(v16bf a, v16bf b, v8f c) {
  return __builtin_amdgcn_wmma_f32_16x16x32_bf16(
      /*neg_a=*/false, a, /*neg_b=*/false, b,
      /*c_mod=*/(short)0, c, /*reuse_a=*/false, /*reuse_b=*/false);
}

// A operand: 16x32 tile from a row-major bf16 matrix [rows, ld] at (r0, k0).
// ISA layout: lanes 0-15 row M=lane, K={0..7,16..23}; lanes 16-31 same rows,
// K={8..15,24..31}.  Each half is 8 contiguous bf16 -> one b128 load.
__device__ __forceinline__ v16bf load_a16x32(const __bf16* base, int ld,
                                             int r0, int k0, int lane) {
  int m  = lane & 15;
  int kb = (lane >> 4) << 3;            // 0 or 8
  const __bf16* p = base + (size_t)(r0 + m) * ld + k0 + kb;
  v8bf lo = *reinterpret_cast<const v8bf*>(p);       // K = kb .. kb+7
  v8bf hi = *reinterpret_cast<const v8bf*>(p + 16);  // K = kb+16 .. kb+23
  v16bf a;
#pragma unroll
  for (int i = 0; i < 8; ++i) { a[i] = lo[i]; a[i + 8] = hi[i]; }
  return a;
}

// B operand: 32x16 tile whose COLUMNS are stored as rows of a transposed,
// row-major bf16 matrix baseT [ncols, ld] (row n = original column n).
// ISA layout: lanes 0-15 col N=lane K={0..7,16..23}; lanes 16-31 same cols K={8..15,24..31}.
__device__ __forceinline__ v16bf load_b32x16(const __bf16* baseT, int ld,
                                             int n0, int k0, int lane) {
  int n  = lane & 15;
  int kb = (lane >> 4) << 3;            // 0 or 8
  const __bf16* p = baseT + (size_t)(n0 + n) * ld + k0 + kb;
  v8bf lo = *reinterpret_cast<const v8bf*>(p);
  v8bf hi = *reinterpret_cast<const v8bf*>(p + 16);
  v16bf b;
#pragma unroll
  for (int i = 0; i < 8; ++i) { b[i] = lo[i]; b[i + 8] = hi[i]; }
  return b;
}

// C/D layout: VGPR j of lanes 0-15 = (M=j, N=lane); lanes 16-31 = (M=j+8).
__device__ __forceinline__ void store_d16x16(float* base, int ld,
                                             int r0, int c0, v8f d, int lane) {
  int mrow = (lane >> 4) << 3;
  int n    = lane & 15;
  float* p = base + (size_t)(r0 + mrow) * ld + c0 + n;
#pragma unroll
  for (int j = 0; j < 8; ++j) p[(size_t)j * ld] = d[j];
}

// ---------------------------------------------------------------------------
// Elementwise / layernorm kernels (256 threads = one D-row per block)
// ---------------------------------------------------------------------------
__global__ void embed_ln_kernel(const int* __restrict__ idx,
                                const float* __restrict__ wte,
                                float* __restrict__ v) {
  int row = blockIdx.x, d = threadIdx.x;
  float x = wte[(size_t)idx[row] * DD + d];
  __shared__ float red[DD];
  red[d] = x; __syncthreads();
  for (int s = 128; s > 0; s >>= 1) { if (d < s) red[d] += red[d + s]; __syncthreads(); }
  float mu = red[0] * (1.0f / DD);
  __syncthreads();
  float c = x - mu;
  red[d] = c * c; __syncthreads();
  for (int s = 128; s > 0; s >>= 1) { if (d < s) red[d] += red[d + s]; __syncthreads(); }
  float var = red[0] * (1.0f / DD);
  v[(size_t)row * DD + d] = c * rsqrtf(var + 1e-5f);
}

__global__ void ln_to_bf_kernel(const float* __restrict__ in,
                                __bf16* __restrict__ out) {
  int row = blockIdx.x, d = threadIdx.x;
  float x = in[(size_t)row * DD + d];
  __shared__ float red[DD];
  red[d] = x; __syncthreads();
  for (int s = 128; s > 0; s >>= 1) { if (d < s) red[d] += red[d + s]; __syncthreads(); }
  float mu = red[0] * (1.0f / DD);
  __syncthreads();
  float c = x - mu;
  red[d] = c * c; __syncthreads();
  for (int s = 128; s > 0; s >>= 1) { if (d < s) red[d] += red[d + s]; __syncthreads(); }
  float var = red[0] * (1.0f / DD);
  out[(size_t)row * DD + d] = (__bf16)(c * rsqrtf(var + 1e-5f));
}

__global__ void add_ln_kernel(float* __restrict__ v,
                              const float* __restrict__ upd) {
  int row = blockIdx.x, d = threadIdx.x;
  float x = upd[(size_t)row * DD + d];
  __shared__ float red[DD];
  red[d] = x; __syncthreads();
  for (int s = 128; s > 0; s >>= 1) { if (d < s) red[d] += red[d + s]; __syncthreads(); }
  float mu = red[0] * (1.0f / DD);
  __syncthreads();
  float c = x - mu;
  red[d] = c * c; __syncthreads();
  for (int s = 128; s > 0; s >>= 1) { if (d < s) red[d] += red[d + s]; __syncthreads(); }
  float var = red[0] * (1.0f / DD);
  v[(size_t)row * DD + d] += c * rsqrtf(var + 1e-5f);
}

// RoPE on v (Q==K==V==v_ast) -> bf16 row-major [B*T, D]
__global__ void rope_kernel(const float* __restrict__ v,
                            __bf16* __restrict__ out) {
  int row = blockIdx.x, d = threadIdx.x;
  int t = row & (TT - 1);
  int j = d & 127;
  // inv_freq = 10000^(-(2j)/256)
  float inv = expf(-(float)(2 * j) * (9.210340371976184f / 256.0f));
  float ang = (float)t * inv;
  float sn, cs;
  sincosf(ang, &sn, &cs);
  float x  = v[(size_t)row * DD + d];
  float xr = (d < 128) ? -v[(size_t)row * DD + d + 128]
                       :  v[(size_t)row * DD + d - 128];
  out[(size_t)row * DD + d] = (__bf16)(x * cs + xr * sn);
}

// f32 -> bf16 elementwise
__global__ void cconv_kernel(const float* __restrict__ in,
                             __bf16* __restrict__ out, int n) {
  int i = blockIdx.x * blockDim.x + threadIdx.x;
  if (i < n) out[i] = (__bf16)in[i];
}

// batched transpose+convert: in f32 [grid.y][R, C] -> out bf16 [grid.y][C, R]
__global__ void tconv_kernel(const float* __restrict__ in,
                             __bf16* __restrict__ out, int R, int C) {
  size_t boff = (size_t)blockIdx.y * R * C;
  int i = blockIdx.x * blockDim.x + threadIdx.x;
  if (i < R * C) {
    int r = i / C, c = i % C;
    out[boff + (size_t)c * R + r] = (__bf16)in[boff + i];
  }
}

// ---------------------------------------------------------------------------
// Causal linear attention (RoPE already applied), unnormalized, flash-style.
// One wave per (batch, 16-query tile); out tile [16 x 256] in f32 registers.
// ---------------------------------------------------------------------------
__global__ __launch_bounds__(32)
void attn_kernel(const __bf16* __restrict__ rope,  // [B, T, D] bf16 (Qr == Kr)
                 const __bf16* __restrict__ vT,    // [B, D, T] bf16
                 float* __restrict__ a_out) {      // [B, T, D] f32
  int blk  = blockIdx.x;
  int b    = blk / (TT / 16);
  int qt   = blk % (TT / 16);
  int lane = threadIdx.x;

  const __bf16* Q  = rope + (size_t)b * TT * DD;
  const __bf16* Kr = Q;                              // K rows == Bᵀ columns
  const __bf16* Vt = vT + (size_t)b * DD * TT;
  float* Ob        = a_out + (size_t)b * TT * DD;

  v8f acc[16];
#pragma unroll
  for (int i = 0; i < 16; ++i) acc[i] = v8f{0, 0, 0, 0, 0, 0, 0, 0};

  __shared__ __attribute__((aligned(16))) __bf16 etile[16 * 32];

  int mrow = (lane >> 4) << 3;
  int ncol = lane & 15;
  int smax = qt * 16 + 15;

  for (int s0 = 0; s0 <= smax; s0 += 32) {
    // Stage 1: energy E[16 x 32] = Qr_tile @ Kr_chunkᵀ  (K = D = 256)
#pragma unroll
    for (int nt = 0; nt < 2; ++nt) {
      v8f e = v8f{0, 0, 0, 0, 0, 0, 0, 0};
#pragma unroll
      for (int kc = 0; kc < 8; ++kc) {
        v16bf a  = load_a16x32(Q, DD, qt * 16, kc * 32, lane);
        v16bf bb = load_b32x16(Kr, DD, s0 + nt * 16, kc * 32, lane);
        e = wmma_bf16(a, bb, e);
      }
      // causal mask + restage to LDS as bf16
      int scol = s0 + nt * 16 + ncol;
#pragma unroll
      for (int j = 0; j < 8; ++j) {
        int trow = qt * 16 + mrow + j;
        float val = (scol <= trow) ? e[j] : 0.0f;
        etile[(mrow + j) * 32 + nt * 16 + ncol] = (__bf16)val;
      }
    }
    __syncthreads();
    // Stage 2: acc[16 x 256] += E @ V_chunk   (K = 32 keys)
    v16bf ea = load_a16x32(etile, 32, 0, 0, lane);
#pragma unroll
    for (int dt = 0; dt < 16; ++dt) {
      v16bf bb = load_b32x16(Vt, TT, dt * 16, s0, lane);
      acc[dt] = wmma_bf16(ea, bb, acc[dt]);
    }
    __syncthreads();
  }
#pragma unroll
  for (int dt = 0; dt < 16; ++dt)
    store_d16x16(Ob, DD, qt * 16, dt * 16, acc[dt], lane);
}

// ---------------------------------------------------------------------------
// Fused MLP: update[16 x 256] = (relu(v@dec_x) * relu(ln_a@dec_y)) @ encoder
// One wave per 16-row tile; N = 8192 streamed in chunks of 32 without ever
// materializing the [B,T,N] tensor.
// ---------------------------------------------------------------------------
__global__ __launch_bounds__(32)
void mlp_fused_kernel(const __bf16* __restrict__ v_bf,   // [B*T, D]
                      const __bf16* __restrict__ lna_bf, // [B*T, D]
                      const __bf16* __restrict__ dxT,    // [H, M, D]
                      const __bf16* __restrict__ dyT,    // [H, M, D]
                      const __bf16* __restrict__ encT,   // [D, N]
                      float* __restrict__ upd) {         // [B*T, D]
  int rt   = blockIdx.x;        // 16-row tile over B*T
  int lane = threadIdx.x;
  int mrow = (lane >> 4) << 3;
  int ncol = lane & 15;

  v8f acc[16];
#pragma unroll
  for (int i = 0; i < 16; ++i) acc[i] = v8f{0, 0, 0, 0, 0, 0, 0, 0};

  __shared__ __attribute__((aligned(16))) __bf16 ytile[16 * 32];

  for (int c = 0; c < NNN; c += 32) {
    int h  = c / MM;
    int m0 = c % MM;
    const __bf16* dx = dxT + (size_t)h * MM * DD;
    const __bf16* dy = dyT + (size_t)h * MM * DD;
#pragma unroll
    for (int nt = 0; nt < 2; ++nt) {
      v8f ex = v8f{0, 0, 0, 0, 0, 0, 0, 0};
      v8f ey = v8f{0, 0, 0, 0, 0, 0, 0, 0};
#pragma unroll
      for (int kc = 0; kc < 8; ++kc) {
        v16bf va = load_a16x32(v_bf,   DD, rt * 16, kc * 32, lane);
        v16bf ya = load_a16x32(lna_bf, DD, rt * 16, kc * 32, lane);
        v16bf bx = load_b32x16(dx, DD, m0 + nt * 16, kc * 32, lane);
        v16bf by = load_b32x16(dy, DD, m0 + nt * 16, kc * 32, lane);
        ex = wmma_bf16(va, bx, ex);
        ey = wmma_bf16(ya, by, ey);
      }
#pragma unroll
      for (int j = 0; j < 8; ++j) {
        float xv = fmaxf(ex[j], 0.0f);
        float yv = fmaxf(ey[j], 0.0f);
        ytile[(mrow + j) * 32 + nt * 16 + ncol] = (__bf16)(xv * yv);
      }
    }
    __syncthreads();
    v16bf yA = load_a16x32(ytile, 32, 0, 0, lane);
#pragma unroll
    for (int dt = 0; dt < 16; ++dt) {
      v16bf be = load_b32x16(encT, NNN, dt * 16, c, lane);
      acc[dt] = wmma_bf16(yA, be, acc[dt]);
    }
    __syncthreads();
  }
#pragma unroll
  for (int dt = 0; dt < 16; ++dt)
    store_d16x16(upd, DD, rt * 16, dt * 16, acc[dt], lane);
}

// ---------------------------------------------------------------------------
// Readout: logits[B*T, 256] = v_bf @ readout   (roT = readoutᵀ, bf16)
// ---------------------------------------------------------------------------
__global__ __launch_bounds__(32)
void readout_kernel(const __bf16* __restrict__ v_bf,
                    const __bf16* __restrict__ roT,   // [V, D]
                    float* __restrict__ out) {
  int rt = blockIdx.x, ct = blockIdx.y, lane = threadIdx.x;
  v8f acc = v8f{0, 0, 0, 0, 0, 0, 0, 0};
#pragma unroll
  for (int kc = 0; kc < 8; ++kc) {
    v16bf a  = load_a16x32(v_bf, DD, rt * 16, kc * 32, lane);
    v16bf bb = load_b32x16(roT, DD, ct * 16, kc * 32, lane);
    acc = wmma_bf16(a, bb, acc);
  }
  store_d16x16(out, VV, rt * 16, ct * 16, acc, lane);
}

// ---------------------------------------------------------------------------
// Host launcher
// ---------------------------------------------------------------------------
extern "C" void kernel_launch(void* const* d_in, const int* in_sizes, int n_in,
                              void* d_out, int out_size, void* d_ws, size_t ws_size,
                              hipStream_t stream) {
  const int*   idx = (const int*)d_in[0];
  const float* wte = (const float*)d_in[1];
  const float* enc = (const float*)d_in[2];
  const float* dx  = (const float*)d_in[3];
  const float* dy  = (const float*)d_in[4];
  const float* ro  = (const float*)d_in[5];
  float*       out = (float*)d_out;

  // ---- workspace carve (256B aligned) ----
  char* ws = (char*)d_ws;
  size_t off = 0;
  auto carve = [&](size_t bytes) {
    char* p = ws + off;
    off += (bytes + 255) & ~(size_t)255;
    return p;
  };
  float*  v_ast   = (float*)carve((size_t)ROWS * DD * 4);
  float*  update  = (float*)carve((size_t)ROWS * DD * 4);
  float*  a_ast   = (float*)carve((size_t)ROWS * DD * 4);
  __bf16* v_bf    = (__bf16*)carve((size_t)ROWS * DD * 2);
  __bf16* vT      = (__bf16*)carve((size_t)BB * DD * TT * 2);
  __bf16* rope_bf = (__bf16*)carve((size_t)ROWS * DD * 2);
  __bf16* lna_bf  = (__bf16*)carve((size_t)ROWS * DD * 2);
  __bf16* dxT     = (__bf16*)carve((size_t)HH * MM * DD * 2);
  __bf16* dyT     = (__bf16*)carve((size_t)HH * MM * DD * 2);
  __bf16* encT    = (__bf16*)carve((size_t)DD * NNN * 2);
  __bf16* roT     = (__bf16*)carve((size_t)VV * DD * 2);
  if (off > ws_size) return;  // insufficient scratch; nothing safe to do

  const int BTD = ROWS * DD;

  // ---- one-time weight transposes to bf16 (B-operand friendly layouts) ----
  tconv_kernel<<<dim3((DD * MM + 255) / 256, HH), 256, 0, stream>>>(dx, dxT, DD, MM);
  tconv_kernel<<<dim3((DD * MM + 255) / 256, HH), 256, 0, stream>>>(dy, dyT, DD, MM);
  tconv_kernel<<<dim3((NNN * DD + 255) / 256, 1), 256, 0, stream>>>(enc, encT, NNN, DD);
  tconv_kernel<<<dim3((DD * VV + 255) / 256, 1), 256, 0, stream>>>(ro, roT, DD, VV);

  // ---- embedding + layernorm ----
  embed_ln_kernel<<<ROWS, DD, 0, stream>>>(idx, wte, v_ast);

  // ---- layers ----
  for (int l = 0; l < LL; ++l) {
    rope_kernel<<<ROWS, DD, 0, stream>>>(v_ast, rope_bf);
    cconv_kernel<<<(BTD + 255) / 256, 256, 0, stream>>>(v_ast, v_bf, BTD);
    tconv_kernel<<<dim3((TT * DD + 255) / 256, BB), 256, 0, stream>>>(v_ast, vT, TT, DD);
    attn_kernel<<<BB * (TT / 16), 32, 0, stream>>>(rope_bf, vT, a_ast);
    ln_to_bf_kernel<<<ROWS, DD, 0, stream>>>(a_ast, lna_bf);
    mlp_fused_kernel<<<ROWS / 16, 32, 0, stream>>>(v_bf, lna_bf, dxT, dyT, encT, update);
    add_ln_kernel<<<ROWS, DD, 0, stream>>>(v_ast, update);
  }

  // ---- readout ----
  cconv_kernel<<<(BTD + 255) / 256, 256, 0, stream>>>(v_ast, v_bf, BTD);
  readout_kernel<<<dim3(ROWS / 16, VV / 16), 32, 0, stream>>>(v_bf, roT, out);
}
